// CiBabyMambaHar_4544075399321
// MI455X (gfx1250) — compile-verified, matrix operand order checked by reading, not run on hardware
//
#include <hip/hip_runtime.h>
#include <hip/hip_bf16.h>

typedef __attribute__((ext_vector_type(16))) _Float16 v16h;
typedef __attribute__((ext_vector_type(8)))  float    v8f;

// Model constants
#define NB   256
#define NC   9
#define NT   128
#define ND   24
#define NS   16
#define NDI  48
#define NPAT 29
#define NSEQ (NB*NC)          // 2304
#define NROW (NSEQ*NPAT)      // 66816  (= 4176 * 16, exact WMMA M-tiles)

// ---------------------------------------------------------------------------
// WMMA operand helpers.  Per the gfx1250 16-bit A/B layout, lane L holds two
// contiguous 8-element K-runs of its row/column:  [kb + 8*(L/16), +8) and
// [kb + 16 + 8*(L/16), +8).  Row strides (24/48 f32) keep all runs 16B-aligned,
// so each run is two global_load_b128.  Runs in the K-pad region are zeroed.
__device__ inline v16h pack16(const float4& r0, const float4& r1,
                              const float4& r2, const float4& r3) {
  v16h a;
  a[0]=(_Float16)r0.x;  a[1]=(_Float16)r0.y;  a[2]=(_Float16)r0.z;  a[3]=(_Float16)r0.w;
  a[4]=(_Float16)r1.x;  a[5]=(_Float16)r1.y;  a[6]=(_Float16)r1.z;  a[7]=(_Float16)r1.w;
  a[8]=(_Float16)r2.x;  a[9]=(_Float16)r2.y;  a[10]=(_Float16)r2.z; a[11]=(_Float16)r2.w;
  a[12]=(_Float16)r3.x; a[13]=(_Float16)r3.y; a[14]=(_Float16)r3.z; a[15]=(_Float16)r3.w;
  return a;
}

// K = 24 (padded to 32): run1 = [8*half,+8) always valid; run2 = [16+8*half,+8)
// valid only for half==0.
__device__ inline v16h load_k24(const float* __restrict__ rp, int half) {
  const float4* p = (const float4*)(rp + 8*half);
  float4 r0 = p[0], r1 = p[1];
  float4 r2 = make_float4(0.f,0.f,0.f,0.f), r3 = r2;
  if (half == 0) {
    const float4* q = (const float4*)(rp + 16);
    r2 = q[0]; r3 = q[1];
  }
  return pack16(r0, r1, r2, r3);
}

// K = 48 (two k-steps of 32): run1 = [ks*32+8*half,+8) always valid (max 47);
// run2 = [ks*32+16+8*half,+8) valid only for ks==0.
__device__ inline v16h load_k48(const float* __restrict__ rp, int half, int ks) {
  const float4* p = (const float4*)(rp + ks*32 + 8*half);
  float4 r0 = p[0], r1 = p[1];
  float4 r2 = make_float4(0.f,0.f,0.f,0.f), r3 = r2;
  if (ks == 0) {
    const float4* q = (const float4*)(rp + 16 + 8*half);
    r2 = q[0]; r3 = q[1];
  }
  return pack16(r0, r1, r2, r3);
}

// ---------------------------------------------------------------------------
// Stem: shared 1-channel conv k=5 pad=2 + BN + SiLU.  h1[n,d,t]
__global__ void k_stem(const float* __restrict__ x, const float* __restrict__ w,
                       const float* __restrict__ g, const float* __restrict__ b,
                       const float* __restrict__ m, const float* __restrict__ v,
                       float* __restrict__ h1) {
  __shared__ float s[NT];
  int n = blockIdx.x, t = threadIdx.x;
  int bi = n / NC, ch = n % NC;
  s[t] = x[(size_t)bi*NT*NC + (size_t)t*NC + ch];
  __syncthreads();
  for (int d = 0; d < ND; ++d) {
    float acc = 0.f;
#pragma unroll
    for (int k = 0; k < 5; ++k) {
      int tt = t + k - 2;
      if (tt >= 0 && tt < NT) acc += w[d*5+k]*s[tt];
    }
    float bn = (acc - m[d]) * rsqrtf(v[d]+1e-5f) * g[d] + b[d];
    h1[(size_t)n*ND*NT + (size_t)d*NT + t] = bn / (1.f + __expf(-bn));
  }
}

// ---------------------------------------------------------------------------
// Patch embed: depthwise conv k=16 s=4 pad=4 + pointwise + BN + pos. h[n,p,d]
__global__ void k_patch(const float* __restrict__ h1, const float* __restrict__ pdw,
                        const float* __restrict__ ppw, const float* __restrict__ g,
                        const float* __restrict__ bb, const float* __restrict__ m,
                        const float* __restrict__ v, const float* __restrict__ pos,
                        float* __restrict__ h) {
  int n = blockIdx.x, p = threadIdx.x;
  if (p >= NPAT) return;
  float sdw[ND];
  const float* base = h1 + (size_t)n*ND*NT;
  for (int d = 0; d < ND; ++d) {
    float acc = 0.f;
#pragma unroll
    for (int k = 0; k < 16; ++k) {
      int idx = p*4 + k - 4;
      if (idx >= 0 && idx < NT) acc += pdw[d*16+k]*base[(size_t)d*NT+idx];
    }
    sdw[d] = acc;
  }
  float* out = h + ((size_t)n*NPAT + p)*ND;
  for (int o = 0; o < ND; ++o) {
    float acc = 0.f;
    for (int d = 0; d < ND; ++d) acc += ppw[o*ND+d]*sdw[d];
    acc = (acc - m[o]) * rsqrtf(v[o]+1e-5f) * g[o] + bb[o];
    out[o] = acc + pos[p*ND+o];
  }
}

// ---------------------------------------------------------------------------
// LayerNorm over last dim (24), one thread per row.
__global__ void k_ln(const float* __restrict__ in, float* __restrict__ out,
                     const float* __restrict__ g, const float* __restrict__ b,
                     int rows) {
  int r = blockIdx.x*blockDim.x + threadIdx.x;
  if (r >= rows) return;
  const float* p = in + (size_t)r*ND;
  float mu = 0.f;
  for (int d = 0; d < ND; ++d) mu += p[d];
  mu *= (1.f/ND);
  float var = 0.f;
  for (int d = 0; d < ND; ++d) { float t = p[d]-mu; var += t*t; }
  var *= (1.f/ND);
  float rs = rsqrtf(var + 1e-5f);
  float* o = out + (size_t)r*ND;
  for (int d = 0; d < ND; ++d) o[d] = (p[d]-mu)*rs*g[d] + b[d];
}

// ---------------------------------------------------------------------------
// in_proj WMMA:  xz[R,96] = hn[R,24] @ in_w[96,24]^T   (K=24 padded to 32)
__global__ void k_inproj(const float* __restrict__ X, const float* __restrict__ W,
                         float* __restrict__ Y) {
  int wave = (blockIdx.x*blockDim.x + threadIdx.x) >> 5;
  int lane = threadIdx.x & 31;
  const int MT = NROW/16, NTILES = 6;
  if (wave >= MT*NTILES) return;                 // wave-uniform, EXEC stays all-1
  int tm = wave / NTILES, tn = wave % NTILES;
  int mrow = lane & 15, half = lane >> 4;
  const float* ap = X + (size_t)(tm*16 + mrow)*ND;       // A row (M = lane%16)
  const float* bp = W + (size_t)(tn*16 + mrow)*ND;       // B col (N = lane%16)
  v16h a  = load_k24(ap, half);
  v16h bm = load_k24(bp, half);
  v8f c = {};
  c = __builtin_amdgcn_wmma_f32_16x16x32_f16(false, a, false, bm, (short)0, c, false, false);
  float* yp = Y + (size_t)(tm*16 + half*8)*96 + tn*16 + mrow;
#pragma unroll
  for (int r = 0; r < 8; ++r) yp[(size_t)r*96] = c[r];
}

// ---------------------------------------------------------------------------
// Depthwise causal conv (k=4) + bias + SiLU.  rev=0 causal fwd, rev=1 anti-causal.
__global__ void k_conv(const float* __restrict__ xz, const float* __restrict__ cw,
                       const float* __restrict__ cb, float* __restrict__ uc, int rev) {
  int idx = blockIdx.x*blockDim.x + threadIdx.x;
  if (idx >= NROW*NDI) return;
  int d = idx % NDI; int rl = idx / NDI; int l = rl % NPAT; int n = rl / NPAT;
  const float* up = xz + (size_t)n*NPAT*96 + d;          // u part = cols [0,48)
  float acc = cb[d];
#pragma unroll
  for (int k = 0; k < 4; ++k) {
    int lp = rev ? (l + 3 - k) : (l - 3 + k);
    if (lp >= 0 && lp < NPAT) acc += cw[d*4+k]*up[(size_t)lp*96];
  }
  uc[(size_t)rl*NDI + d] = acc / (1.f + __expf(-acc));
}

// ---------------------------------------------------------------------------
// x_proj WMMA:  dbl[R,48] (cols 0..33 valid) = uc[R,48] @ xproj_w[34,48]^T
// Layout of dbl row: [0:2) = dtr, [2:18) = B, [18:34) = C.
__global__ void k_xproj(const float* __restrict__ X, const float* __restrict__ W,
                        float* __restrict__ dbl) {
  int wave = (blockIdx.x*blockDim.x + threadIdx.x) >> 5;
  int lane = threadIdx.x & 31;
  const int MT = NROW/16, NTILES = 3;
  if (wave >= MT*NTILES) return;
  int tm = wave / NTILES, tn = wave % NTILES;
  int mrow = lane & 15, half = lane >> 4;
  int col = tn*16 + mrow;
  bool bvalid = (col < 34);
  int bcol = bvalid ? col : 0;
  const float* ap = X + (size_t)(tm*16 + mrow)*NDI;
  const float* bp = W + (size_t)bcol*NDI;
  v16h zero16 = {};
  v8f c = {};
#pragma unroll
  for (int ks = 0; ks < 2; ++ks) {
    v16h a  = load_k48(ap, half, ks);
    v16h bm = load_k48(bp, half, ks);
    if (!bvalid) bm = zero16;
    c = __builtin_amdgcn_wmma_f32_16x16x32_f16(false, a, false, bm, (short)0, c, false, false);
  }
  if (bvalid) {
    float* dp = dbl + (size_t)(tm*16 + half*8)*NDI + col;
#pragma unroll
    for (int r = 0; r < 8; ++r) dp[(size_t)r*NDI] = c[r];
  }
}

// ---------------------------------------------------------------------------
// dt head: dt = softplus(dtr @ dt_w^T + dt_b), one thread per (row,d).
__global__ void k_dthead(const float* __restrict__ dbl, const float* __restrict__ dtw,
                         const float* __restrict__ dtb, float* __restrict__ dts) {
  int idx = blockIdx.x*blockDim.x + threadIdx.x;
  if (idx >= NROW*NDI) return;
  int d = idx % NDI; int row = idx / NDI;
  float d0 = dbl[(size_t)row*NDI + 0];
  float d1 = dbl[(size_t)row*NDI + 1];
  float xv = d0*dtw[d*2] + d1*dtw[d*2+1] + dtb[d];
  dts[(size_t)row*NDI + d] = (xv > 20.f) ? xv : log1pf(__expf(xv));
}

// ---------------------------------------------------------------------------
// Selective scan + skip (u*Dp) + SiLU(z) gate.  One thread per (n,d), S=16 in regs.
// B/C read directly from dbl rows (offsets 2 / 18, stride 48).
__global__ void k_scan(const float* __restrict__ uc, const float* __restrict__ dts,
                       const float* __restrict__ dbl, const float* __restrict__ xz,
                       const float* __restrict__ Alog, const float* __restrict__ Dpw,
                       float* __restrict__ y, int rev) {
  int idx = blockIdx.x*blockDim.x + threadIdx.x;
  if (idx >= NSEQ*NDI) return;
  int d = idx % NDI; int n = idx / NDI;
  float A[NS], hst[NS];
#pragma unroll
  for (int s = 0; s < NS; ++s) { A[s] = -__expf(Alog[d*NS+s]); hst[s] = 0.f; }
  float dpd = Dpw[d];
  for (int t = 0; t < NPAT; ++t) {
    int l = rev ? (NPAT-1-t) : t;
    size_t row = (size_t)n*NPAT + l;
    float dt = dts[row*NDI+d];
    float uv = uc[row*NDI+d];
    float du = dt*uv;
    const float* bp = dbl + row*NDI + 2;
    const float* cp = dbl + row*NDI + 18;
    float acc = 0.f;
#pragma unroll
    for (int s = 0; s < NS; ++s) {
      hst[s] = __expf(dt*A[s])*hst[s] + du*bp[s];
      acc += hst[s]*cp[s];
    }
    float zv = xz[row*96 + NDI + d];
    y[row*NDI+d] = (acc + uv*dpd) * (zv/(1.f+__expf(-zv)));
  }
}

// ---------------------------------------------------------------------------
// out_proj WMMA with residual accumulate:  Dst[R,24] = Csrc + Yin[R,48] @ W[24,48]^T
__global__ void k_outproj(const float* __restrict__ Yin, const float* __restrict__ W,
                          const float* __restrict__ Csrc, float* __restrict__ Dst) {
  int wave = (blockIdx.x*blockDim.x + threadIdx.x) >> 5;
  int lane = threadIdx.x & 31;
  const int MT = NROW/16, NTILES = 2;
  if (wave >= MT*NTILES) return;
  int tm = wave / NTILES, tn = wave % NTILES;
  int mrow = lane & 15, half = lane >> 4;
  int col = tn*16 + mrow;
  bool bvalid = (col < ND);
  int bcol = bvalid ? col : 0;
  // seed accumulator with the residual (C input of WMMA)
  v8f c;
  const float* cpb = Csrc + (size_t)(tm*16 + half*8)*ND;
#pragma unroll
  for (int r = 0; r < 8; ++r) c[r] = bvalid ? cpb[(size_t)r*ND + col] : 0.f;
  const float* ap = Yin + (size_t)(tm*16 + mrow)*NDI;
  const float* bp = W + (size_t)bcol*NDI;
  v16h zero16 = {};
#pragma unroll
  for (int ks = 0; ks < 2; ++ks) {                       // K = 48 -> 2 x 32
    v16h a  = load_k48(ap, half, ks);
    v16h bm = load_k48(bp, half, ks);
    if (!bvalid) bm = zero16;
    c = __builtin_amdgcn_wmma_f32_16x16x32_f16(false, a, false, bm, (short)0, c, false, false);
  }
  if (bvalid) {
    float* dp = Dst + (size_t)(tm*16 + half*8)*ND + col;
#pragma unroll
    for (int r = 0; r < 8; ++r) dp[(size_t)r*ND] = c[r];
  }
}

// ---------------------------------------------------------------------------
// Gated temporal attention pooling per sequence.
__global__ void k_attnpool(const float* __restrict__ h, const float* __restrict__ aw,
                           const float* __restrict__ ab, const float* __restrict__ ctx,
                           float* __restrict__ cpool) {
  __shared__ float sc[NPAT];
  __shared__ float alpha[NPAT];
  int n = blockIdx.x, tid = threadIdx.x;
  const float* hb = h + (size_t)n*NPAT*ND;
  if (tid < NPAT) {
    float hr[ND];
    for (int d = 0; d < ND; ++d) hr[d] = hb[(size_t)tid*ND+d];
    float s = 0.f;
    for (int d = 0; d < ND; ++d) {
      float u = ab[d];
      for (int e = 0; e < ND; ++e) u += aw[d*ND+e]*hr[e];
      s += tanhf(u)*ctx[d];
    }
    sc[tid] = s;
  }
  __syncthreads();
  if (tid == 0) {
    float mx = sc[0];
    for (int l = 1; l < NPAT; ++l) mx = fmaxf(mx, sc[l]);
    float sum = 0.f;
    for (int l = 0; l < NPAT; ++l) { alpha[l] = __expf(sc[l]-mx); sum += alpha[l]; }
    float inv = 1.f/sum;
    for (int l = 0; l < NPAT; ++l) alpha[l] *= inv;
  }
  __syncthreads();
  if (tid < ND) {
    float acc = 0.f;
    for (int l = 0; l < NPAT; ++l) acc += alpha[l]*hb[(size_t)l*ND+tid];
    cpool[(size_t)n*ND+tid] = acc;
  }
}

// ---------------------------------------------------------------------------
// Channel fusion mean + head LN + classifier.
__global__ void k_head(const float* __restrict__ cpool, const float* __restrict__ g,
                       const float* __restrict__ b, const float* __restrict__ hw,
                       const float* __restrict__ hbias, float* __restrict__ out) {
  __shared__ float cm[ND];
  __shared__ float stats[2];
  int bi = blockIdx.x, tid = threadIdx.x;
  if (tid < ND) {
    float acc = 0.f;
    for (int ch = 0; ch < NC; ++ch) acc += cpool[((size_t)bi*NC+ch)*ND+tid];
    cm[tid] = acc * (1.f/NC);
  }
  __syncthreads();
  if (tid == 0) {
    float mu = 0.f;
    for (int d = 0; d < ND; ++d) mu += cm[d];
    mu *= (1.f/ND);
    float var = 0.f;
    for (int d = 0; d < ND; ++d) { float t = cm[d]-mu; var += t*t; }
    var *= (1.f/ND);
    stats[0] = mu; stats[1] = rsqrtf(var+1e-5f);
  }
  __syncthreads();
  if (tid < 6) {
    float acc = hbias[tid];
    for (int d = 0; d < ND; ++d)
      acc += hw[tid*ND+d]*((cm[d]-stats[0])*stats[1]*g[d]+b[d]);
    out[(size_t)bi*6+tid] = acc;
  }
}

// ---------------------------------------------------------------------------
extern "C" void kernel_launch(void* const* d_in, const int* in_sizes, int n_in,
                              void* d_out, int out_size, void* d_ws, size_t ws_size,
                              hipStream_t stream) {
  const float* x         = (const float*)d_in[0];
  const float* stem_w    = (const float*)d_in[1];
  const float* stem_bn_g = (const float*)d_in[2];
  const float* stem_bn_b = (const float*)d_in[3];
  const float* stem_bn_m = (const float*)d_in[4];
  const float* stem_bn_v = (const float*)d_in[5];
  const float* pd_w      = (const float*)d_in[6];
  const float* pp_w      = (const float*)d_in[7];
  const float* pat_bn_g  = (const float*)d_in[8];
  const float* pat_bn_b  = (const float*)d_in[9];
  const float* pat_bn_m  = (const float*)d_in[10];
  const float* pat_bn_v  = (const float*)d_in[11];
  const float* pos_embed = (const float*)d_in[12];
  const float* ln1_g     = (const float*)d_in[13];
  const float* ln1_b     = (const float*)d_in[14];
  const float* ln2_g     = (const float*)d_in[15];
  const float* ln2_b     = (const float*)d_in[16];
  const float* in_w      = (const float*)d_in[17];
  const float* conv_w    = (const float*)d_in[18];
  const float* conv_b    = (const float*)d_in[19];
  const float* xproj_w   = (const float*)d_in[20];
  const float* dt_w      = (const float*)d_in[21];
  const float* dt_bias   = (const float*)d_in[22];
  const float* A_log     = (const float*)d_in[23];
  const float* Dp        = (const float*)d_in[24];
  const float* out_w     = (const float*)d_in[25];
  const float* attn_w    = (const float*)d_in[26];
  const float* attn_b    = (const float*)d_in[27];
  const float* ctx       = (const float*)d_in[28];
  const float* head_g    = (const float*)d_in[29];
  const float* head_b    = (const float*)d_in[30];
  const float* head_w    = (const float*)d_in[31];
  const float* head_bias = (const float*)d_in[32];
  float* out = (float*)d_out;

  // Workspace layout (floats); every buffer offset is a multiple of R floats
  // (R*4 bytes = 267264, divisible by 16) so all float4 accesses stay aligned.
  float* W  = (float*)d_ws;
  const size_t R = NROW;
  float* h    = W;                    // R*24
  float* hn   = h   + R*ND;           // R*24
  float* tmp  = hn  + R*ND;           // R*24
  float* uc   = tmp + R*ND;           // R*48
  float* dts  = uc  + R*NDI;          // R*48
  float* dbl  = dts + R*NDI;          // R*48  (dtr | B | C, cols 0..33 valid)
  float* ybuf = dbl + R*NDI;          // R*48
  float* scratch = ybuf + R*NDI;      // max(h1: 2304*24*128, xz: R*96) -- aliased
  float* h1 = scratch;
  float* xz = scratch;
  float* cpool = scratch + (size_t)NSEQ*ND*NT;  // 2304*24 (used after layers)

  // Stem + patch embed
  k_stem<<<NSEQ, NT, 0, stream>>>(x, stem_w, stem_bn_g, stem_bn_b, stem_bn_m,
                                  stem_bn_v, h1);
  k_patch<<<NSEQ, 32, 0, stream>>>(h1, pd_w, pp_w, pat_bn_g, pat_bn_b, pat_bn_m,
                                   pat_bn_v, pos_embed, h);

  const int lnBlocks   = (int)((R + 255) / 256);
  const int ipBlocks   = (NROW/16)*6/8;          // 8 waves per 256-thread block
  const int convBlocks = (NROW*NDI + 255) / 256;
  const int xpBlocks   = ((NROW/16)*3 + 7)/8;
  const int dtBlocks   = (NROW*NDI + 255) / 256;
  const int scBlocks   = (NSEQ*NDI + 255) / 256;
  const int opBlocks   = (NROW/16)*2/8;

  for (int i = 0; i < 4; ++i) {
    k_ln<<<lnBlocks, 256, 0, stream>>>(h, hn, ln1_g + i*ND, ln1_b + i*ND, (int)R);
    k_inproj<<<ipBlocks, 256, 0, stream>>>(hn, in_w + (size_t)i*96*ND, xz);
    for (int dir = 0; dir < 2; ++dir) {
      k_conv<<<convBlocks, 256, 0, stream>>>(xz, conv_w + (size_t)i*NDI*4,
                                             conv_b + (size_t)i*NDI, uc, dir);
      k_xproj<<<xpBlocks, 256, 0, stream>>>(uc, xproj_w + (size_t)i*34*NDI, dbl);
      k_dthead<<<dtBlocks, 256, 0, stream>>>(dbl, dt_w + (size_t)i*NDI*2,
                                             dt_bias + (size_t)i*NDI, dts);
      k_scan<<<scBlocks, 256, 0, stream>>>(uc, dts, dbl, xz,
                                           A_log + (size_t)i*NDI*NS,
                                           Dp + (size_t)i*NDI, ybuf, dir);
      // dir 0: tmp = h + y@W^T ; dir 1: tmp += y@W^T  (residual folded into C)
      k_outproj<<<opBlocks, 256, 0, stream>>>(ybuf, out_w + (size_t)i*ND*NDI,
                                              (dir == 0) ? h : tmp, tmp);
    }
    k_ln<<<lnBlocks, 256, 0, stream>>>(tmp, h, ln2_g + i*ND, ln2_b + i*ND, (int)R);
  }

  k_attnpool<<<NSEQ, 32, 0, stream>>>(h, attn_w, attn_b, ctx, cpool);
  k_head<<<NB, 32, 0, stream>>>(cpool, head_g, head_b, head_w, head_bias, out);
}
